// APPNPModel_79121887527327
// MI455X (gfx1250) — compile-verified, matrix operand order checked by reading, not run on hardware
//
#include <hip/hip_runtime.h>

typedef __attribute__((ext_vector_type(16))) _Float16 v16h;
typedef __attribute__((ext_vector_type(8)))  _Float16 v8h;
typedef __attribute__((ext_vector_type(8)))  float    v8f;

static constexpr int   N_NODES = 100000;
static constexpr int   E_EDGES = 1000000;
static constexpr int   D_IN    = 256;
static constexpr int   D_HID   = 256;
static constexpr int   D_OUT   = 64;
static constexpr int   K_STEPS = 10;
static constexpr float ALPHA   = 0.1f;

// ---------------------------------------------------------------------------
// fp32 -> f16 conversion (grid-stride)
// ---------------------------------------------------------------------------
__global__ void cvt_f32_to_f16(const float* __restrict__ in,
                               _Float16* __restrict__ out, int n) {
  for (int i = blockIdx.x * blockDim.x + threadIdx.x; i < n;
       i += gridDim.x * blockDim.x)
    out[i] = (_Float16)in[i];
}

// W [K,N] row-major (fp32) -> Wt [N,K] row-major (f16)
__global__ void transpose_to_f16(const float* __restrict__ W,
                                 _Float16* __restrict__ Wt, int K, int N) {
  int i = blockIdx.x * blockDim.x + threadIdx.x;
  if (i >= K * N) return;
  int k = i / N, n = i % N;
  Wt[(size_t)n * K + k] = (_Float16)W[(size_t)k * N + n];
}

// ---------------------------------------------------------------------------
// WMMA GEMM: out = relu(A @ B + bias); A [M,KDIM] f16 row-major,
// Bt [NCOLS,KDIM] f16 (B transposed). One wave -> one 16x16 tile.
// ---------------------------------------------------------------------------
template <int NCOLS, int KDIM, bool OUT_F16>
__global__ __launch_bounds__(256) void gemm_relu_wmma(
    const _Float16* __restrict__ A, const _Float16* __restrict__ Bt,
    const float* __restrict__ bias, void* __restrict__ out, int M) {
  const int lane = threadIdx.x & 31;          // wave32
  const int wid  = blockIdx.x * 8 + (threadIdx.x >> 5);
  constexpr int tiles_n = NCOLS / 16;
  const int tile_m = wid / tiles_n;
  const int tile_n = wid % tiles_n;
  if (tile_m * 16 >= M) return;               // wave-uniform: EXEC stays all-1s

  const int sub = lane >> 4;                  // 0 or 1 (half-wave)
  const int l15 = lane & 15;

  // A fragment: lane holds row (tile_m*16 + l15); per 32-K step the 16 halves
  // are two contiguous 8-half runs at k0+sub*8 and k0+16+sub*8 -> 2x b128.
  const _Float16* arow = A  + (size_t)(tile_m * 16 + l15) * KDIM;
  // B fragment: lane holds column (tile_n*16 + l15) of B = row of Bt.
  const _Float16* brow = Bt + (size_t)(tile_n * 16 + l15) * KDIM;
  const float bval = bias[tile_n * 16 + l15];

  v8f c = {};
#pragma unroll
  for (int k0 = 0; k0 < KDIM; k0 += 32) {
    v16h a, b;
    ((v8h*)&a)[0] = *(const v8h*)(arow + k0 + sub * 8);
    ((v8h*)&a)[1] = *(const v8h*)(arow + k0 + 16 + sub * 8);
    ((v8h*)&b)[0] = *(const v8h*)(brow + k0 + sub * 8);
    ((v8h*)&b)[1] = *(const v8h*)(brow + k0 + 16 + sub * 8);
    if (k0 + 64 < KDIM) {
      __builtin_prefetch(arow + k0 + 64, 0, 0);
      __builtin_prefetch(brow + k0 + 64, 0, 0);
    }
    c = __builtin_amdgcn_wmma_f32_16x16x32_f16(
        /*neg_a=*/false, a, /*neg_b=*/false, b,
        /*c_mod=*/(short)0, c, /*reuse_a=*/false, /*reuse_b=*/false);
  }

  // C/D layout: VGPR r -> M = tile_m*16 + sub*8 + r, N = tile_n*16 + l15
#pragma unroll
  for (int r = 0; r < 8; ++r) {
    const size_t m = (size_t)(tile_m * 16 + sub * 8 + r);
    const int    n = tile_n * 16 + l15;
    float v = c[r] + bval;
    v = v > 0.0f ? v : 0.0f;
    if constexpr (OUT_F16)
      ((_Float16*)out)[m * NCOLS + n] = (_Float16)v;
    else
      ((float*)out)[m * NCOLS + n] = v;
  }
}

// ---------------------------------------------------------------------------
// APPNP propagation helpers
// ---------------------------------------------------------------------------
__device__ __forceinline__ void atomic_fadd(float* p, float v) {
  __hip_atomic_fetch_add(p, v, __ATOMIC_RELAXED, __HIP_MEMORY_SCOPE_AGENT);
}

__global__ void init_deg(float* __restrict__ deg) {
  int i = blockIdx.x * blockDim.x + threadIdx.x;
  if (i < N_NODES) deg[i] = 1.0f;  // self-loop
}

__global__ void accum_deg(const int* __restrict__ dst, float* __restrict__ deg) {
  int e = blockIdx.x * blockDim.x + threadIdx.x;
  if (e < E_EDGES) atomic_fadd(&deg[dst[e]], 1.0f);
}

__global__ void calc_dinv(const float* __restrict__ deg,
                          float* __restrict__ dinv) {
  int i = blockIdx.x * blockDim.x + threadIdx.x;
  if (i < N_NODES) {
    float d = deg[i];
    dinv[i] = d > 0.0f ? rsqrtf(d) : 0.0f;
  }
}

__global__ void calc_norm(const int* __restrict__ src,
                          const int* __restrict__ dst,
                          const float* __restrict__ dinv,
                          float* __restrict__ norm) {
  int e = blockIdx.x * blockDim.x + threadIdx.x;
  if (e < E_EDGES) norm[e] = dinv[src[e]] * dinv[dst[e]];
}

__global__ void zero_f4(float4* __restrict__ p, int n4) {
  int i = blockIdx.x * blockDim.x + threadIdx.x;
  if (i < n4) p[i] = make_float4(0.f, 0.f, 0.f, 0.f);
}

// One wave per edge; lane handles features {lane, lane+32}. Gather z[src],
// scale by norm, scatter-add into agg[dst] with hw f32 atomics.
__global__ __launch_bounds__(256) void scatter_edges(
    const int* __restrict__ src, const int* __restrict__ dst,
    const float* __restrict__ norm, const float* __restrict__ z,
    float* __restrict__ agg) {
  const int lane = threadIdx.x & 31;
  const int e = blockIdx.x * 8 + (threadIdx.x >> 5);
  if (e >= E_EDGES) return;
  const int s = src[e], d = dst[e];
  const float w = norm[e];
  const float* zs = z + (size_t)s * D_OUT;
  float* ad = agg + (size_t)d * D_OUT;
  atomic_fadd(ad + lane,      zs[lane] * w);
  atomic_fadd(ad + lane + 32, zs[lane + 32] * w);
}

// z_new = (1-a)*(agg + dinv^2 * z) + a*h   (self-loop folded in)
__global__ void combine_step(const float* __restrict__ agg,
                             const float* __restrict__ z,
                             const float* __restrict__ h,
                             const float* __restrict__ dinv,
                             float* __restrict__ out, int total) {
  int i = blockIdx.x * blockDim.x + threadIdx.x;
  if (i >= total) return;
  int v = i >> 6;  // D_OUT == 64
  float sn = dinv[v];
  sn *= sn;
  float a = agg[i] + z[i] * sn;
  out[i] = (1.0f - ALPHA) * a + ALPHA * h[i];
}

// ---------------------------------------------------------------------------
extern "C" void kernel_launch(void* const* d_in, const int* in_sizes, int n_in,
                              void* d_out, int out_size, void* d_ws,
                              size_t ws_size, hipStream_t stream) {
  const float* x   = (const float*)d_in[0];
  const int*   ei  = (const int*)d_in[1];
  const float* W1  = (const float*)d_in[2];
  const float* b1  = (const float*)d_in[3];
  const float* W2  = (const float*)d_in[4];
  const float* b2  = (const float*)d_in[5];
  const int* src = ei;
  const int* dst = ei + E_EDGES;

  // workspace carve-out (256B aligned slices)
  char* ws = (char*)d_ws;
  size_t off = 0;
  auto take = [&](size_t bytes) {
    char* p = ws + off;
    off += (bytes + 255) & ~(size_t)255;
    return p;
  };
  _Float16* x16  = (_Float16*)take((size_t)N_NODES * D_IN * 2);   // dead after GEMM1
  _Float16* h1   = (_Float16*)take((size_t)N_NODES * D_HID * 2);  // dead after GEMM2
  _Float16* w1t  = (_Float16*)take((size_t)D_HID * D_IN * 2);
  _Float16* w2t  = (_Float16*)take((size_t)D_OUT * D_HID * 2);
  float*    h2   = (float*)take((size_t)N_NODES * D_OUT * 4);
  float*    agg  = (float*)take((size_t)N_NODES * D_OUT * 4);
  float*    deg  = (float*)take((size_t)N_NODES * 4);
  float*    dinv = (float*)take((size_t)N_NODES * 4);
  float*    nrm  = (float*)take((size_t)E_EDGES * 4);
  float* zA = (float*)x16;  // reuse retired f16 buffers as z ping-pong
  float* zB = (float*)h1;

  // 1) f16 staging for the WMMA GEMMs
  cvt_f32_to_f16<<<4096, 256, 0, stream>>>(x, x16, N_NODES * D_IN);
  transpose_to_f16<<<(D_IN * D_HID + 255) / 256, 256, 0, stream>>>(W1, w1t,
                                                                   D_IN, D_HID);
  transpose_to_f16<<<(D_HID * D_OUT + 255) / 256, 256, 0, stream>>>(W2, w2t,
                                                                    D_HID, D_OUT);

  // 2) MLP: 100000 = 6250 full 16-row tiles, no tail
  gemm_relu_wmma<D_HID, D_IN, true>
      <<<(N_NODES / 16) * (D_HID / 16) / 8, 256, 0, stream>>>(x16, w1t, b1, h1,
                                                              N_NODES);
  gemm_relu_wmma<D_OUT, D_HID, false>
      <<<(N_NODES / 16) * (D_OUT / 16) / 8, 256, 0, stream>>>(h1, w2t, b2, h2,
                                                              N_NODES);

  // 3) GCN normalization
  init_deg<<<(N_NODES + 255) / 256, 256, 0, stream>>>(deg);
  accum_deg<<<(E_EDGES + 255) / 256, 256, 0, stream>>>(dst, deg);
  calc_dinv<<<(N_NODES + 255) / 256, 256, 0, stream>>>(deg, dinv);
  calc_norm<<<(E_EDGES + 255) / 256, 256, 0, stream>>>(src, dst, dinv, nrm);

  // 4) K propagation steps; last one writes d_out directly
  const int total = N_NODES * D_OUT;
  const float* zcur = h2;
  for (int k = 0; k < K_STEPS; ++k) {
    float* zout = (k == K_STEPS - 1) ? (float*)d_out : ((k & 1) ? zB : zA);
    zero_f4<<<(total / 4 + 255) / 256, 256, 0, stream>>>((float4*)agg,
                                                         total / 4);
    scatter_edges<<<E_EDGES / 8, 256, 0, stream>>>(src, dst, nrm, zcur, agg);
    combine_step<<<(total + 255) / 256, 256, 0, stream>>>(agg, zcur, h2, dinv,
                                                          zout, total);
    zcur = zout;
  }
}
